// GaussianAttention_51230369906982
// MI455X (gfx1250) — matrix operand order repaired and checked by reading, not compiled, not run on hardware
//
#include <hip/hip_runtime.h>
#include <hip/hip_bf16.h>
#include <math.h>

// ---------------------------------------------------------------------------
// GaussianAttention for MI455X (gfx1250, wave32, WMMA f16 16x16x32, fp32 acc)
// B=4, C=256, H=W=64 -> N=4096, CQ=64, RED=16
// q/k/v/W are stored in WMMA-fragment-major layouts so every matrix operand is
// loaded as a contiguous 32B-per-lane chunk (global_load_b128 pairs).
// ---------------------------------------------------------------------------

#define BATCH 4
#define CCH   256          // channels
#define CQD   64           // C/4
#define NPIX  4096         // H*W
#define CCHUNK 128         // channel chunk per attention block (2 chunks)

typedef _Float16 half16 __attribute__((ext_vector_type(16)));
typedef float    float8 __attribute__((ext_vector_type(8)));

__device__ __forceinline__ float8 wmma_f16(half16 a, half16 b, float8 c) {
    return __builtin_amdgcn_wmma_f32_16x16x32_f16(false, a, false, b,
                                                  (short)0, c, false, false);
}

// ---- WMMA fragment layouts (as used here, contiguous k; 16-bit A 16x32 / B 32x16) ----------
// A-frag: lane = m + 16*hi; lane's halfs i cover k_local = 16*hi + i
// B-frag: lane = n + 16*hi; lane's halfs i cover k_local = 16*hi + i

// ---------------------------------------------------------------------------
// K0: zero the attention column-sum accumulator (needed every launch)
// ---------------------------------------------------------------------------
__global__ void ga_zero_kernel(float* __restrict__ p, int n) {
    int i = blockIdx.x * blockDim.x + threadIdx.x;
    if (i < n) p[i] = 0.0f;
}

// ---------------------------------------------------------------------------
// K0b: scatter a weight matrix W[K, Ncols] (f32) into B-frag-major f16:
//      wf[(kt*(Ncols/16) + nt)*32 + lane][16],  kt = k>>5
// ---------------------------------------------------------------------------
__global__ void ga_wfrag_kernel(const float* __restrict__ W,
                                _Float16* __restrict__ wf, int Ncols, int total) {
    int idx = blockIdx.x * blockDim.x + threadIdx.x;
    if (idx >= total) return;
    unsigned k = (unsigned)idx / (unsigned)Ncols;
    unsigned n = (unsigned)idx % (unsigned)Ncols;
    unsigned kl = k & 31u;
    unsigned lane = (n & 15u) + 16u * (kl >> 4);
    unsigned o = (((k >> 5) * (unsigned)(Ncols >> 4) + (n >> 4)) * 32u + lane) * 16u + (kl & 15u);
    wf[o] = (_Float16)W[idx];
}

// ---------------------------------------------------------------------------
// K1: projection GEMM  out[b, m, n] = (sum_k x[b, k, m] * W[k, n] + bias[n])*sc
//     x is [B, 256, NPIX]; W pre-scattered into B-frag f16.
//     MODE 0: out -> A-frag tiles  [b][m>>4][k/32][lane][16]   (q, sc=1/8)
//     MODE 1: out -> B-frag tiles  [b][m>>4][k/32][lane][16]   (k, for QK^T)
//     MODE 2: out -> B-frag tiles  [b][m>>5][n>>4][lane][16]   (v, for PV)
// ---------------------------------------------------------------------------
template <int MODE>
__global__ void ga_proj_kernel(const float* __restrict__ x,
                               const _Float16* __restrict__ wf,
                               const float* __restrict__ bias,
                               _Float16* __restrict__ out,
                               int K, int Ncols, float out_scale) {
    const int tilesN = Ncols >> 4;
    const unsigned tm = (unsigned)blockIdx.x / (unsigned)tilesN;
    const unsigned tn = (unsigned)blockIdx.x % (unsigned)tilesN;
    const unsigned b = blockIdx.y;
    const unsigned lane = threadIdx.x;
    const unsigned hi = lane >> 4;

    const unsigned m = (tm << 4) + (lane & 15u);
    const unsigned n = (tn << 4) + (lane & 15u);

    // x pointer: covers k = 16*hi + i; advance 32 rows per iteration
    const float* xp = x + (size_t)b * (256u * NPIX) + m + hi * (16u * NPIX);
    const half16* wv = reinterpret_cast<const half16*>(wf);
    const half16* wp = wv + tn * 32u + lane;        // advance tilesN*32 per iter

    float8 acc = {};
    for (int k0 = 0; k0 < K; k0 += 32) {
        half16 af;
#pragma unroll
        for (int i = 0; i < 16; ++i)
            af[i] = (_Float16)xp[(unsigned)i * NPIX];
        acc = wmma_f16(af, wp[0], acc);
        xp += 32u * NPIX;
        wp += (unsigned)tilesN * 32u;
    }
    const float bn = bias[n];
#pragma unroll
    for (int r = 0; r < 8; ++r) {
        const unsigned row = (tm << 4) + (unsigned)r + 8u * hi;  // pixel index
        const unsigned col = n;                                  // channel
        const _Float16 val = (_Float16)((acc[r] + bn) * out_scale);
        unsigned o;
        if (MODE == 0) {            // q -> A-frag order
            unsigned cl = col & 31u;
            unsigned h2 = cl >> 4;
            unsigned ii = cl & 15u;
            o = (((b * (NPIX / 16) + (row >> 4)) * 2u + (col >> 5)) * 32u +
                 ((row & 15u) + 16u * h2)) * 16u + ii;
        } else if (MODE == 1) {     // k -> B-frag order (16-row key tiles)
            unsigned lane2 = (row & 15u) + 16u * ((col >> 4) & 1u);
            o = (((b * (NPIX / 16) + (row >> 4)) * 2u + (col >> 5)) * 32u + lane2)
                    * 16u + (col & 15u);
        } else {                    // v -> B-frag order (32-row key tiles)
            unsigned kl = row & 31u;
            unsigned lane2 = (col & 15u) + 16u * (kl >> 4);
            o = (((b * (NPIX / 32) + (row >> 5)) * (unsigned)(Ncols >> 4) +
                  (col >> 4)) * 32u + lane2) * 16u + (kl & 15u);
        }
        out[o] = val;
    }
}

// ---------------------------------------------------------------------------
// K2: softmax stats pass. One wave per 16-query tile. Per-lane online (m,l)
//     over that lane's column subset; single 16-lane merge at the end.
//     q is pre-scaled by 1/8, so S needs no extra scaling.
// ---------------------------------------------------------------------------
__global__ void ga_stats_kernel(const _Float16* __restrict__ qh,
                                const _Float16* __restrict__ kh,
                                float* __restrict__ m_buf,
                                float* __restrict__ linv_buf) {
    const unsigned qt = blockIdx.x;
    const unsigned b = blockIdx.y;
    const unsigned lane = threadIdx.x;
    const unsigned hi = lane >> 4;

    const half16* qv = reinterpret_cast<const half16*>(qh);
    const half16* kp = reinterpret_cast<const half16*>(kh) +
                       b * (unsigned)(NPIX / 16) * 64u + lane;
    const half16* qp = qv + (b * (unsigned)(NPIX / 16) + qt) * 64u + lane;

    half16 qa0 = qp[0], qa1 = qp[32];

    float lm[8], ll[8];
#pragma unroll
    for (int r = 0; r < 8; ++r) { lm[r] = -1e30f; ll[r] = 0.0f; }

    for (int kt = 0; kt < NPIX / 16; ++kt) {
        __builtin_prefetch((const void*)(kp + 64), 0, 1);
        float8 S = {};
        S = wmma_f16(qa0, kp[0], S);
        S = wmma_f16(qa1, kp[32], S);
        kp += 64;
#pragma unroll
        for (int r = 0; r < 8; ++r) {
            float v = S[r];
            float nm = fmaxf(lm[r], v);
            ll[r] = ll[r] * __expf(lm[r] - nm) + __expf(v - nm);
            lm[r] = nm;
        }
    }
    // merge the 16 per-lane partials within each half-group
#pragma unroll
    for (int r = 0; r < 8; ++r) {
        float m = lm[r], l = ll[r];
        for (int off = 1; off < 16; off <<= 1) {
            float om = __shfl_xor(m, off, 32);
            float ol = __shfl_xor(l, off, 32);
            float nm = fmaxf(m, om);
            l = l * __expf(m - nm) + ol * __expf(om - nm);
            m = nm;
        }
        lm[r] = m; ll[r] = l;
    }
    if ((lane & 15u) == 0) {
#pragma unroll
        for (int r = 0; r < 8; ++r) {
            unsigned row = (qt << 4) + (unsigned)r + 8u * hi;
            m_buf[b * NPIX + row] = lm[r];
            linv_buf[b * NPIX + row] = 1.0f / ll[r];
        }
    }
}

// ---------------------------------------------------------------------------
// K3: attention pass. Recompute S, form P*D = exp(S - m - gauss)/l in regs
//     (mask fused into the softmax exponential), scatter to A-frag order in
//     LDS, WMMA against V fragments. att column sums from c-chunk 0 only.
// ---------------------------------------------------------------------------
__global__ void ga_attn_kernel(const _Float16* __restrict__ qh,
                               const _Float16* __restrict__ kh,
                               const _Float16* __restrict__ vh,
                               const float* __restrict__ m_buf,
                               const float* __restrict__ linv_buf,
                               const float* __restrict__ log_sigma_sq,
                               float* __restrict__ att,
                               float* __restrict__ out_ws) {
    const unsigned qt = blockIdx.x;
    const unsigned cc = blockIdx.y;          // channel chunk (CCHUNK wide)
    const unsigned b = blockIdx.z;
    const unsigned lane = threadIdx.x;
    const unsigned hi = lane >> 4;
    // combined Gaussian coefficient: exp(-(dx^2+dy^2) * gcoef)
    const float gcoef = (1.0f / 7938.0f) * 0.5f / __expf(log_sigma_sq[0]);

    __shared__ __align__(32) _Float16 ptile[32 * 16]; // P*D in A-frag order

    const half16* qp = reinterpret_cast<const half16*>(qh) +
                       (b * (unsigned)(NPIX / 16) + qt) * 64u + lane;
    const half16* kp = reinterpret_cast<const half16*>(kh) +
                       b * (unsigned)(NPIX / 16) * 64u + lane;
    const half16* vp = reinterpret_cast<const half16*>(vh) +
                       b * (unsigned)(NPIX / 32) * 512u + cc * (CCHUNK / 16) * 32u + lane;

    half16 qa0 = qp[0], qa1 = qp[32];

    float rm[8], rl[8];
    int qpx[8], qpy[8];
#pragma unroll
    for (int r = 0; r < 8; ++r) {
        unsigned row = (qt << 4) + (unsigned)r + 8u * hi;
        rm[r] = m_buf[b * NPIX + row];
        rl[r] = linv_buf[b * NPIX + row];
        qpx[r] = (int)(row >> 6);
        qpy[r] = (int)(row & 63u);
    }

    float8 acc[CCHUNK / 16];
#pragma unroll
    for (int nt = 0; nt < CCHUNK / 16; ++nt) acc[nt] = (float8){};

    for (int kt = 0; kt < NPIX / 32; ++kt) {
        __builtin_prefetch((const void*)(kp + 128), 0, 1);
        __builtin_prefetch((const void*)(vp + 512), 0, 1);

        // --- build P*D tile (16 queries x 32 keys), two 16-key subtiles ---
#pragma unroll
        for (int ks = 0; ks < 2; ++ks) {
            const int key = kt * 32 + ks * 16 + (int)(lane & 15u);
            float8 S = {};
            S = wmma_f16(qa0, kp[ks * 64 + 0], S);
            S = wmma_f16(qa1, kp[ks * 64 + 32], S);

            const int kpx = key >> 6, kpy = key & 63;
            const unsigned hi_r = (unsigned)ks;
            const unsigned ii = (lane & 15u);
            float csum = 0.0f;
#pragma unroll
            for (int r = 0; r < 8; ++r) {
                int dx = qpx[r] - kpx, dy = qpy[r] - kpy;
                float garg = (float)(dx * dx + dy * dy) * gcoef;
                float pd = __expf(S[r] - rm[r] - garg) * rl[r];
                csum += pd;
                // scatter into A-frag order: m = r+8*hi, k_local = ks*16+(lane&15)
                unsigned lane_r = (unsigned)r + 8u * hi + 16u * hi_r;
                ptile[lane_r * 16u + ii] = (_Float16)pd;
            }
            if (cc == 0) {
                csum += __shfl_xor(csum, 16, 32);
                if (lane < 16) atomicAdd(&att[b * NPIX + (unsigned)key], csum);
            }
        }
        kp += 128;
        __syncthreads();

        // --- P (16x32 f16 A-frag, contiguous LDS read) @ V (32 x CCHUNK) ---
        half16 pa = *reinterpret_cast<const half16*>(&ptile[lane * 16u]);
#pragma unroll
        for (int nt = 0; nt < CCHUNK / 16; ++nt)
            acc[nt] = wmma_f16(pa, vp[nt * 32], acc[nt]);
        vp += 512;
        __syncthreads();
    }

    float* ob = out_ws + (size_t)b * NPIX * CCH;
#pragma unroll
    for (int nt = 0; nt < CCHUNK / 16; ++nt) {
        const unsigned col = cc * CCHUNK + (unsigned)nt * 16u + (lane & 15u);
#pragma unroll
        for (int r = 0; r < 8; ++r) {
            unsigned row = (qt << 4) + (unsigned)r + 8u * hi;
            ob[row * CCH + col] = acc[nt][r];
        }
    }
}

// ---------------------------------------------------------------------------
// K4: per-batch centroid + Gaussian SE + gate MLP. One 256-thread block/batch.
// ---------------------------------------------------------------------------
__device__ __forceinline__ float ga_block_reduce(float v, float* red, int t) {
    red[t] = v; __syncthreads();
    for (int s = 128; s > 0; s >>= 1) {
        if (t < s) red[t] += red[t + s];
        __syncthreads();
    }
    float r = red[0]; __syncthreads();
    return r;
}

__global__ void ga_se_kernel(const float* __restrict__ att,
                             const float* __restrict__ out_ws,
                             const float* __restrict__ log_sigma_se,
                             const float* __restrict__ W1,
                             const float* __restrict__ W2,
                             float* __restrict__ gate) {
    const int b = blockIdx.x;
    const int t = threadIdx.x;
    __shared__ float red[256];
    __shared__ float kern[NPIX];
    __shared__ float sq[CCH];
    __shared__ float hid[16];

    // centroid from attention column sums
    float s = 0.0f, sx = 0.0f, sy = 0.0f;
    for (int i = t; i < NPIX; i += 256) {
        float a = att[(size_t)b * NPIX + i];
        s += a;
        sx += (float)(i >> 6) * a;
        sy += (float)(i & 63) * a;
    }
    float stot = ga_block_reduce(s, red, t);
    float sxt  = ga_block_reduce(sx, red, t);
    float syt  = ga_block_reduce(sy, red, t);
    const float mu_h = sxt / ((stot + 1e-8f) * 63.0f);
    const float mu_w = syt / ((stot + 1e-8f) * 63.0f);

    // Gaussian SE kernel, staged in LDS
    const float inv2 = 0.5f / __expf(log_sigma_se[0]);
    float ks = 0.0f;
    for (int i = t; i < NPIX; i += 256) {
        float gx = (float)(i >> 6) * (1.0f / 63.0f);
        float gy = (float)(i & 63) * (1.0f / 63.0f);
        float dx = gx - mu_h, dy = gy - mu_w;
        float e = __expf(-(dx * dx + dy * dy) * inv2);
        kern[i] = e;
        ks += e;
    }
    float ktot = ga_block_reduce(ks, red, t);
    const float inv_ks = 1.0f / ktot;
    __syncthreads();

    // squeezed[c] = sum_hw out[b,hw,c] * kern[hw]/ktot  (c = thread id)
    float accu = 0.0f;
    const float* ob = out_ws + (size_t)b * NPIX * CCH;
    for (int hw = 0; hw < NPIX; ++hw)
        accu += ob[(size_t)hw * CCH + t] * kern[hw];
    sq[t] = accu * inv_ks;
    __syncthreads();

    // gate = sigmoid(relu(sq @ W1) @ W2)
    if (t < 16) {
        float h = 0.0f;
        for (int c = 0; c < CCH; ++c) h += sq[c] * W1[c * 16 + t];
        hid[t] = fmaxf(h, 0.0f);
    }
    __syncthreads();
    float g = 0.0f;
#pragma unroll
    for (int j = 0; j < 16; ++j) g += hid[j] * W2[j * CCH + t];
    gate[(size_t)b * CCH + t] = 1.0f / (1.0f + __expf(-g));
}

// ---------------------------------------------------------------------------
// K5: final  d_out[b,c,h,w] = out_ws[b, hw, c] * gate[b,c]
// ---------------------------------------------------------------------------
__global__ void ga_final_kernel(const float* __restrict__ out_ws,
                                const float* __restrict__ gate,
                                float* __restrict__ out) {
    const unsigned idx = blockIdx.x * blockDim.x + threadIdx.x; // < 2^22
    const unsigned b = idx >> 20;           // C*HW = 2^20
    const unsigned c = (idx >> 12) & 255u;  // HW = 2^12
    const unsigned hw = idx & 4095u;
    out[idx] = out_ws[(b << 20) + (hw << 8) + c] * gate[b * CCH + c];
}

// ---------------------------------------------------------------------------
extern "C" void kernel_launch(void* const* d_in, const int* in_sizes, int n_in,
                              void* d_out, int out_size, void* d_ws, size_t ws_size,
                              hipStream_t stream) {
    const float* x_Q  = (const float*)d_in[0];
    const float* x_KV = (const float*)d_in[1];
    const float* Wq   = (const float*)d_in[2];
    const float* bq   = (const float*)d_in[3];
    const float* Wk   = (const float*)d_in[4];
    const float* bk   = (const float*)d_in[5];
    const float* Wv   = (const float*)d_in[6];
    const float* bv   = (const float*)d_in[7];
    const float* lssq = (const float*)d_in[8];
    const float* lsse = (const float*)d_in[9];
    const float* W1   = (const float*)d_in[10];
    const float* W2   = (const float*)d_in[11];
    float* out = (float*)d_out;

    // workspace layout (all fragment buffers 32B aligned)
    char* ws = (char*)d_ws;
    size_t off = 0;
    _Float16* qh  = (_Float16*)(ws + off); off += (size_t)BATCH * NPIX * CQD * 2;  // 2 MB
    _Float16* kh  = (_Float16*)(ws + off); off += (size_t)BATCH * NPIX * CQD * 2;  // 2 MB
    _Float16* vh  = (_Float16*)(ws + off); off += (size_t)BATCH * NPIX * CCH * 2;  // 8 MB
    _Float16* wqf = (_Float16*)(ws + off); off += (size_t)CCH * CQD * 2;           // 32 KB
    _Float16* wkf = (_Float16*)(ws + off); off += (size_t)CCH * CQD * 2;           // 32 KB
    _Float16* wvf = (_Float16*)(ws + off); off += (size_t)CCH * CCH * 2;           // 128 KB
    float* m_buf  = (float*)(ws + off);    off += (size_t)BATCH * NPIX * 4;
    float* l_buf  = (float*)(ws + off);    off += (size_t)BATCH * NPIX * 4;
    float* att    = (float*)(ws + off);    off += (size_t)BATCH * NPIX * 4;
    float* ows    = (float*)(ws + off);    off += (size_t)BATCH * NPIX * CCH * 4;  // 16.8 MB
    float* gate   = (float*)(ws + off);    off += (size_t)BATCH * CCH * 4;

    // K0: zero att accumulator; pre-scatter weights into B-frag f16
    ga_zero_kernel<<<(BATCH * NPIX + 255) / 256, 256, 0, stream>>>(att, BATCH * NPIX);
    ga_wfrag_kernel<<<(CCH * CQD + 255) / 256, 256, 0, stream>>>(Wq, wqf, CQD, CCH * CQD);
    ga_wfrag_kernel<<<(CCH * CQD + 255) / 256, 256, 0, stream>>>(Wk, wkf, CQD, CCH * CQD);
    ga_wfrag_kernel<<<(CCH * CCH + 255) / 256, 256, 0, stream>>>(Wv, wvf, CCH, CCH * CCH);

    // K1: projections (WMMA GEMMs), fragment-major f16 outputs
    ga_proj_kernel<0><<<dim3((NPIX / 16) * (CQD / 16), BATCH), 32, 0, stream>>>(
        x_Q, wqf, bq, qh, CCH, CQD, 0.125f);   // q pre-scaled by (C/4)^-0.5
    ga_proj_kernel<1><<<dim3((NPIX / 16) * (CQD / 16), BATCH), 32, 0, stream>>>(
        x_KV, wkf, bk, kh, CCH, CQD, 1.0f);
    ga_proj_kernel<2><<<dim3((NPIX / 16) * (CCH / 16), BATCH), 32, 0, stream>>>(
        x_KV, wvf, bv, vh, CCH, CCH, 1.0f);

    // K2: exact softmax row stats (flash pass 1)
    ga_stats_kernel<<<dim3(NPIX / 16, BATCH), 32, 0, stream>>>(qh, kh, m_buf, l_buf);

    // K3: attention (flash pass 2): P*D -> out, and att column sums
    ga_attn_kernel<<<dim3(NPIX / 16, CCH / CCHUNK, BATCH), 32, 0, stream>>>(
        qh, kh, vh, m_buf, l_buf, lssq, att, ows);

    // K4: centroid + Gaussian SE + gate
    ga_se_kernel<<<BATCH, 256, 0, stream>>>(att, ows, lsse, W1, W2, gate);

    // K5: apply gate, transpose to [B, C, H, W]
    ga_final_kernel<<<(BATCH * CCH * NPIX) / 256, 256, 0, stream>>>(ows, gate, out);
}
